// FIBERCrossModalAttention_64647847739940
// MI455X (gfx1250) — matrix operand order repaired and checked by reading, not compiled, not run on hardware
//
#include <hip/hip_runtime.h>
#include <hip/hip_bf16.h>

// Problem dims (compile-time constants from the reference)
#define B_   8
#define SQ_  1024
#define SKV_ 512
#define QD_  1024
#define KVD_ 768
#define H_   16
#define HD_  64
#define EPS_ 1e-5f
#define SCALE_ 0.125f              // HD^-0.5 = 1/8
#define INV_H_ (1.0f / 16.0f)

typedef __attribute__((ext_vector_type(16))) __bf16 v16bf;
typedef __attribute__((ext_vector_type(8)))  float  v8f;
typedef __attribute__((ext_vector_type(4)))  float  f32x4;

// ---------------------------------------------------------------------------
// WMMA fragment helpers (wave32, V_WMMA_F32_16X16X32_BF16; contiguous k order, not the ISA 7.12.2 table)
// A (16x32): lane lo=row; slot s -> K = 16*hi + s
// B (32x16): lane lo=col; slot s -> K = 16*hi + s
// C/D (16x16 f32): VGPR r, lane: n=lo, m=r+8*hi
// ---------------------------------------------------------------------------
__device__ __forceinline__ v16bf cvt16(f32x4 x0, f32x4 x1, f32x4 x2, f32x4 x3) {
  v16bf r;
#pragma unroll
  for (int i = 0; i < 4; ++i) {
    r[i]      = (__bf16)x0[i];
    r[4 + i]  = (__bf16)x1[i];
    r[8 + i]  = (__bf16)x2[i];
    r[12 + i] = (__bf16)x3[i];
  }
  return r;
}

// Row-major A with contiguous K: 4x b128 loads + pack-convert.
__device__ __forceinline__ v16bf load_a_frag_rm(const float* P, long ld,
                                                int m0, int k0) {
  const int lane = threadIdx.x & 31;
  const int lo = lane & 15, hi = lane >> 4;
  const float* row = P + (long)(m0 + lo) * ld + k0 + 16 * hi;
  f32x4 x0 = *(const f32x4*)(row);
  f32x4 x1 = *(const f32x4*)(row + 4);
  f32x4 x2 = *(const f32x4*)(row + 8);
  f32x4 x3 = *(const f32x4*)(row + 12);
  return cvt16(x0, x1, x2, x3);
}

// B fragment where K is memory-contiguous (e.g. Q@K^T): 4x b128 + convert.
// Element (k, n) = P[n*nstride + k].
__device__ __forceinline__ v16bf load_b_frag_kc(const float* P, long nstride,
                                                int k0, int n0) {
  const int lane = threadIdx.x & 31;
  const int lo = lane & 15, hi = lane >> 4;
  const float* col = P + (long)(n0 + lo) * nstride + k0 + 16 * hi;
  f32x4 x0 = *(const f32x4*)(col);
  f32x4 x1 = *(const f32x4*)(col + 4);
  f32x4 x2 = *(const f32x4*)(col + 8);
  f32x4 x3 = *(const f32x4*)(col + 12);
  return cvt16(x0, x1, x2, x3);
}

// Generic B fragment, K-strided (row-major W[k][n] or V). Scalar loads.
__device__ __forceinline__ v16bf load_b_frag_st(const float* P, long ld,
                                                int k0, int n0) {
  const int lane = threadIdx.x & 31;
  const int lo = lane & 15, hi = lane >> 4;
  const long n = n0 + lo;
  v16bf b;
#pragma unroll
  for (int s = 0; s < 16; ++s)
    b[s] = (__bf16)P[(long)(k0 + 16 * hi + s) * ld + n];
  return b;
}

__device__ __forceinline__ v8f wmma_bf16(v16bf a, v16bf b, v8f c) {
  return __builtin_amdgcn_wmma_f32_16x16x32_bf16(
      false, a, false, b, (short)0, c, false, false);
}

// Per-lane async global->LDS 16-byte copy (gfx1250 ASYNCcnt path).
__device__ __forceinline__ void async_copy_b128(unsigned lds_off, const float* gaddr) {
  asm volatile("global_load_async_to_lds_b128 %0, %1, off"
               :: "v"(lds_off), "v"(gaddr) : "memory");
}
__device__ __forceinline__ void wait_async_all() {
  asm volatile("s_wait_asynccnt 0" ::: "memory");
}
__device__ __forceinline__ unsigned lds_off_of(const void* p) {
  return (unsigned)(size_t)p;   // low 32 bits of generic addr == LDS byte offset
}

// ---------------------------------------------------------------------------
// GEMM + bias:  C[M,N] = A[M,K] @ W[K,N] + bias[N]  (fp32 io, bf16 mul, f32 acc)
// Block = 256 threads (8 waves) -> 128x128 C tile; K stepped by 64 with
// async-staged LDS tiles: A 128x64 f32 (32KB), W 64x128 f32 (32KB).
// Wave w: wave_m = w>>2 (64 rows), wave_n = w&3 (32 cols) -> 4x2 subtiles
// (8 f32x8 accumulators), 16 WMMAs per K-step.
// M multiple of 128, N multiple of 128, K multiple of 64.
// ---------------------------------------------------------------------------
__global__ __launch_bounds__(256)
void gemm_bias_kernel(const float* __restrict__ A, const float* __restrict__ W,
                      const float* __restrict__ bias, float* __restrict__ C,
                      int M, int N, int K) {
  __shared__ __align__(16) float Asm[128 * 64];   // 32 KB
  __shared__ __align__(16) float Wsm[64 * 128];   // 32 KB

  const int tid = threadIdx.x;
  const int wave = tid >> 5;
  const int lane = tid & 31;
  const int lo = lane & 15, hi = lane >> 4;

  const int m_blk = blockIdx.y * 128;
  const int n_blk = blockIdx.x * 128;
  const int wm = (wave >> 2) * 64;       // wave row origin within tile
  const int wn = (wave & 3) * 32;        // wave col origin within tile

  // A-tile chunks (2048 of 16B): row = c/16, col4 = (c%16)*4 ; 8 per thread
  const int ar = (tid >> 4), ac = (tid & 15) * 4;
  // W-tile chunks (2048 of 16B): row = c/32, col4 = (c%32)*4 ; 8 per thread
  const int wr = (tid >> 5), wc = (tid & 31) * 4;

  v8f acc[4][2];
#pragma unroll
  for (int mi = 0; mi < 4; ++mi)
#pragma unroll
    for (int ni = 0; ni < 2; ++ni)
      acc[mi][ni] = (v8f){};

  for (int k0 = 0; k0 < K; k0 += 64) {
    __syncthreads();   // previous iteration's LDS reads complete
#pragma unroll
    for (int i = 0; i < 8; ++i) {        // 8 chunks per thread per tile
      const int arr = ar + i * 16;       // +256 chunks => +16 rows
      async_copy_b128(lds_off_of(&Asm[arr * 64 + ac]),
                      A + (long)(m_blk + arr) * K + k0 + ac);
      const int wrr = wr + i * 8;        // +256 chunks => +8 rows
      async_copy_b128(lds_off_of(&Wsm[wrr * 128 + wc]),
                      W + (long)(k0 + wrr) * N + n_blk + wc);
    }
    wait_async_all();
    __syncthreads();

#pragma unroll
    for (int kk = 0; kk < 64; kk += 32) {
      v16bf a[4];
#pragma unroll
      for (int mi = 0; mi < 4; ++mi)
        a[mi] = load_a_frag_rm(Asm, 64, wm + mi * 16, kk);
#pragma unroll
      for (int ni = 0; ni < 2; ++ni) {
        v16bf b = load_b_frag_st(Wsm, 128, kk, wn + ni * 16);
#pragma unroll
        for (int mi = 0; mi < 4; ++mi)
          acc[mi][ni] = wmma_bf16(a[mi], b, acc[mi][ni]);
      }
    }
  }

#pragma unroll
  for (int mi = 0; mi < 4; ++mi) {
#pragma unroll
    for (int ni = 0; ni < 2; ++ni) {
      const int n = n_blk + wn + ni * 16 + lo;
      const float bv = bias[n];
      const int m0 = m_blk + wm + mi * 16 + 8 * hi;
#pragma unroll
      for (int r = 0; r < 8; ++r)
        C[(long)(m0 + r) * N + n] = acc[mi][ni][r] + bv;
    }
  }
}

// ---------------------------------------------------------------------------
// Attention: one block per (b*h, 32-query-row tile).
//   S[32][512] f32 in LDS (64KB), WMMA scores (vectorized Q / K^T fragments),
//   row softmax, atomic head-mean, WMMA context.
// ---------------------------------------------------------------------------
__global__ __launch_bounds__(256)
void attn_kernel(const float* __restrict__ qp, const float* __restrict__ kp,
                 const float* __restrict__ vp, const float* __restrict__ mask,
                 float* __restrict__ attn_mean, float* __restrict__ ctx) {
  __shared__ __align__(16) float S[32][SKV_];   // 64 KB

  const int bh = blockIdx.x;
  const int b = bh / H_, h = bh % H_;
  const int q0 = blockIdx.y * 32;
  const int wave = threadIdx.x >> 5;
  const int lane = threadIdx.x & 31;
  const int lo = lane & 15, hi = lane >> 4;

  const float* Qp = qp + (long)b * SQ_ * QD_ + (long)h * HD_;  // (m,k): Qp[m*QD+k]
  const float* Kp = kp + (long)b * SKV_ * QD_ + (long)h * HD_; // (k,n): Kp[n*QD+k]
  const float* Vp = vp + (long)b * SKV_ * QD_ + (long)h * HD_; // (k,n): Vp[k*QD+n]

  // Phase 1: S = scale * Q @ K^T + mask-bias. 2x32 grid of 16x16 subtiles.
  for (int s = wave; s < 64; s += 8) {
    const int ms = s >> 5, ns = s & 31;
    v8f acc = {};
#pragma unroll
    for (int k0 = 0; k0 < HD_; k0 += 32) {
      v16bf a  = load_a_frag_rm(Qp, QD_, q0 + ms * 16, k0);
      v16bf bb = load_b_frag_kc(Kp, QD_, k0, ns * 16);   // K^T: k contiguous
      acc = wmma_bf16(a, bb, acc);
    }
    const int n = ns * 16 + lo;
    const float bias = (1.0f - mask[b * SKV_ + n]) * -10000.0f;
#pragma unroll
    for (int r = 0; r < 8; ++r)
      S[ms * 16 + r + 8 * hi][n] = acc[r] * SCALE_ + bias;
  }
  __syncthreads();

  // Phase 2: row softmax + head-mean accumulation (threads 0..31, row each).
  if (threadIdx.x < 32) {
    const int m = threadIdx.x;
    float mx = -1e30f;
    for (int n = 0; n < SKV_; ++n) mx = fmaxf(mx, S[m][n]);
    float sum = 0.0f;
    for (int n = 0; n < SKV_; ++n) {
      const float e = __expf(S[m][n] - mx);
      S[m][n] = e;
      sum += e;
    }
    const float inv = 1.0f / sum;
    float* am = attn_mean + ((long)b * SQ_ + q0 + m) * SKV_;
    for (int n = 0; n < SKV_; ++n) {
      const float p = S[m][n] * inv;
      S[m][n] = p;
      atomicAdd(&am[n], p * INV_H_);
    }
  }
  __syncthreads();

  // Phase 3: ctx = P @ V. 2x4 grid of 16x16 subtiles, one per wave.
  {
    const int ms = wave >> 2, ns = wave & 3;
    v8f acc = {};
    for (int k0 = 0; k0 < SKV_; k0 += 32) {
      v16bf a  = load_a_frag_rm(&S[0][0], SKV_, ms * 16, k0);  // LDS, vectorized
      v16bf bb = load_b_frag_st(Vp, QD_, k0, ns * 16);
      acc = wmma_bf16(a, bb, acc);
    }
    const int n = ns * 16 + lo;
#pragma unroll
    for (int r = 0; r < 8; ++r)
      ctx[((long)b * SQ_ + q0 + ms * 16 + r + 8 * hi) * QD_ + (long)h * HD_ + n] = acc[r];
  }
}

// ---------------------------------------------------------------------------
// fused = LayerNorm(query + alpha * oproj) * gamma + beta.  Block per row,
// float4 per thread (QD = 1024 = 256 threads * 4).
// ---------------------------------------------------------------------------
__global__ __launch_bounds__(256)
void fuse_ln_kernel(const float* __restrict__ query, const float* __restrict__ oproj,
                    const float* __restrict__ alpha, const float* __restrict__ gamma,
                    const float* __restrict__ beta, float* __restrict__ out) {
  __shared__ float red[256];
  const long row = blockIdx.x;
  const int tid = threadIdx.x;
  const int c = tid * 4;
  const float a = alpha[0];

  f32x4 q4 = *(const f32x4*)(query + row * QD_ + c);
  f32x4 o4 = *(const f32x4*)(oproj + row * QD_ + c);
  f32x4 f = q4 + o4 * a;

  red[tid] = f[0] + f[1] + f[2] + f[3];
  __syncthreads();
  for (int s = 128; s > 0; s >>= 1) {
    if (tid < s) red[tid] += red[tid + s];
    __syncthreads();
  }
  const float mean = red[0] * (1.0f / QD_);
  __syncthreads();

  f32x4 d = f - mean;
  red[tid] = d[0] * d[0] + d[1] * d[1] + d[2] * d[2] + d[3] * d[3];
  __syncthreads();
  for (int s = 128; s > 0; s >>= 1) {
    if (tid < s) red[tid] += red[tid + s];
    __syncthreads();
  }
  const float rstd = rsqrtf(red[0] * (1.0f / QD_) + EPS_);

  f32x4 g4 = *(const f32x4*)(gamma + c);
  f32x4 b4 = *(const f32x4*)(beta + c);
  *(f32x4*)(out + row * QD_ + c) = d * rstd * g4 + b4;
}

__global__ void zero_kernel(float* __restrict__ p, long n) {
  const long i = ((long)blockIdx.x * blockDim.x + threadIdx.x) * 4;
  if (i < n) *(f32x4*)(p + i) = (f32x4){0.f, 0.f, 0.f, 0.f};
}

// ---------------------------------------------------------------------------
// Launch
// ---------------------------------------------------------------------------
extern "C" void kernel_launch(void* const* d_in, const int* in_sizes, int n_in,
                              void* d_out, int out_size, void* d_ws, size_t ws_size,
                              hipStream_t stream) {
  const float* query = (const float*)d_in[0];   // [B,SQ,QD]
  const float* keyv  = (const float*)d_in[1];   // [B,SKV,KVD]
  const float* mask  = (const float*)d_in[2];   // [B,SKV]
  const float* Wq    = (const float*)d_in[3];
  const float* bq    = (const float*)d_in[4];
  const float* Wk    = (const float*)d_in[5];
  const float* bk    = (const float*)d_in[6];
  const float* Wv    = (const float*)d_in[7];
  const float* bv    = (const float*)d_in[8];
  const float* Wo    = (const float*)d_in[9];
  const float* bo    = (const float*)d_in[10];
  const float* gamma = (const float*)d_in[11];
  const float* beta  = (const float*)d_in[12];
  const float* alpha = (const float*)d_in[13];

  float* fused     = (float*)d_out;                              // [B,SQ,QD]
  float* attn_mean = fused + (long)B_ * SQ_ * QD_;               // [B,SQ,SKV]

  float* ws = (float*)d_ws;
  float* qp  = ws;                                  // [B*SQ, QD]   8M floats
  float* kp  = qp  + (long)B_ * SQ_  * QD_;         // [B*SKV, QD]  4M
  float* vp  = kp  + (long)B_ * SKV_ * QD_;         // [B*SKV, QD]  4M
  float* ctx = vp  + (long)B_ * SKV_ * QD_;         // [B*SQ, QD]   8M
  float* op  = ctx + (long)B_ * SQ_  * QD_;         // [B*SQ, QD]   8M

  // Projections (128x128 tiles)
  gemm_bias_kernel<<<dim3(QD_ / 128, (B_ * SQ_) / 128), 256, 0, stream>>>(
      query, Wq, bq, qp, B_ * SQ_, QD_, QD_);
  gemm_bias_kernel<<<dim3(QD_ / 128, (B_ * SKV_) / 128), 256, 0, stream>>>(
      keyv, Wk, bk, kp, B_ * SKV_, QD_, KVD_);
  gemm_bias_kernel<<<dim3(QD_ / 128, (B_ * SKV_) / 128), 256, 0, stream>>>(
      keyv, Wv, bv, vp, B_ * SKV_, QD_, KVD_);

  // attn_mean accumulated atomically -> zero first
  {
    const long n = (long)B_ * SQ_ * SKV_;      // multiple of 1024
    zero_kernel<<<(unsigned)(n / 1024), 256, 0, stream>>>(attn_mean, n);
  }

  // Attention
  attn_kernel<<<dim3(B_ * H_, SQ_ / 32), 256, 0, stream>>>(
      qp, kp, vp, mask, attn_mean, ctx);

  // Output projection
  gemm_bias_kernel<<<dim3(QD_ / 128, (B_ * SQ_) / 128), 256, 0, stream>>>(
      ctx, Wo, bo, op, B_ * SQ_, QD_, QD_);

  // Residual + LayerNorm
  fuse_ln_kernel<<<dim3(B_ * SQ_), 256, 0, stream>>>(
      query, op, alpha, gamma, beta, fused);
}
